// ObjectDetector_37280316129899
// MI455X (gfx1250) — compile-verified, matrix operand order checked by reading, not run on hardware
//
#include <hip/hip_runtime.h>

// ---------------- types for WMMA fragments ----------------
typedef __attribute__((ext_vector_type(16))) __bf16       bf16x16;
typedef __attribute__((ext_vector_type(8)))  float        f32x8;
typedef __attribute__((ext_vector_type(16))) short        s16x16;
typedef __attribute__((ext_vector_type(4)))  unsigned int u32x4;

union Frag {
    s16x16  s;
    u32x4   d[2];
    bf16x16 b;
};

__device__ __forceinline__ unsigned short f2bf(float f) {
    unsigned int u = __builtin_bit_cast(unsigned int, f);
    u += 0x7FFFu + ((u >> 16) & 1u);           // round-to-nearest-even
    return (unsigned short)(u >> 16);
}
__device__ __forceinline__ float bf2f(unsigned short h) {
    unsigned int u = ((unsigned int)h) << 16;
    return __builtin_bit_cast(float, u);
}

// ---------------- NCHW f32 -> NHWC bf16 ----------------
__global__ void nchw_to_nhwc(const float* __restrict__ x, unsigned short* __restrict__ y,
                             int N, int C, int H, int W) {
    long idx = (long)blockIdx.x * blockDim.x + threadIdx.x;
    long total = (long)N * C * H * W;
    if (idx >= total) return;
    int w = idx % W;
    long t = idx / W;
    int h = t % H; t /= H;
    int c = t % C;
    int n = t / C;
    y[(((size_t)n * H + h) * W + w) * C + c] = f2bf(x[idx]);
}

// ---------------- weight pack: OIHW f32 -> B-fragment-order bf16 ----------------
// K ordering: cell = kh*KW+kw outer, channel inner, channels padded to Cpad = ceil(Cin/32)*32.
// layout: wpk[((nTile*kChunks + kc)*32 + lane)*16 + slot]
// slot->K: K = kc*32 + (lane>=16 ? 8 : 0) + (slot/8)*16 + (slot%8); col = nTile*16 + lane%16
__global__ void pack_weights(const float* __restrict__ w, unsigned short* __restrict__ wpk,
                             int Cin, int Cout, int KH, int KW, int nTilesAlloc) {
    int idx = blockIdx.x * blockDim.x + threadIdx.x;
    int Cpad = ((Cin + 31) >> 5) << 5;
    int kChunks = (KH * KW * Cpad) >> 5;
    int total = nTilesAlloc * kChunks * 512;
    if (idx >= total) return;
    int slot = idx & 15;
    int lane = (idx >> 4) & 31;
    int g    = idx >> 9;
    int kc   = g % kChunks;
    int nT   = g / kChunks;
    int hi   = lane >> 4;
    int col  = lane & 15;
    int K    = kc * 32 + hi * 8 + (slot < 8 ? slot : 16 + (slot - 8));
    int cell = K / Cpad;
    int ci   = K - cell * Cpad;
    int kw   = cell % KW;
    int kh   = cell / KW;
    int cout = nT * 16 + col;
    unsigned short v = 0;
    if (ci < Cin && cout < Cout)
        v = f2bf(w[(((size_t)cout * Cin + ci) * KH + kh) * KW + kw]);
    wpk[idx] = v;
}

// ---------------- implicit-GEMM conv via WMMA bf16 ----------------
// in: NHWC bf16; wpk: packed fragments; out: NHWC bf16 with channel stride ldC, offset cBase.
// Spatial dims are powers of two: woShift = log2(Wo), hwShift = log2(Ho*Wo).
// STAGE: cooperatively stage the block's whole packed-B region into LDS (8x reuse across waves).
template <int KH, int KW, int STRIDE, int PAD, bool RELU, bool VEC8, int NB, bool STAGE>
__global__ __launch_bounds__(256) void conv_wmma(
    const unsigned short* __restrict__ in,
    const unsigned short* __restrict__ wpk,
    const float* __restrict__ bias,
    unsigned short* __restrict__ out,
    int Nb, int H, int W, int Cin, int Ho, int Wo, int Cout, int ldC, int cBase,
    int woShift, int hwShift)
{
    extern __shared__ unsigned short sB[];
    const int lane = threadIdx.x & 31;
    const int wave = threadIdx.x >> 5;
    const int cc32 = (Cin + 31) >> 5;          // K-chunks per (kh,kw) cell
    const int kChunks = KH * KW * cc32;
    const int nTile0 = blockIdx.y * NB;

    if constexpr (STAGE) {
        // copy [nTile0 .. nTile0+NB) packed-B (contiguous) into LDS as 16B vectors
        const u32x4* gsrc = (const u32x4*)(wpk + (size_t)nTile0 * kChunks * 512);
        u32x4* ldst = (u32x4*)sB;
        int nvec = kChunks * NB * 64;          // 1KB per (kc,nb) = 64 x 16B
        for (int i = threadIdx.x; i < nvec; i += 256) ldst[i] = gsrc[i];
        __syncthreads();
    }

    const int Mtot = Nb * Ho * Wo;
    const int mBase = (blockIdx.x * 8 + wave) * 16;
    if (mBase >= Mtot) return;                 // wave-uniform exit (after barrier)
    const int col = lane & 15;                 // A row select / B+D column select
    const int hi  = lane >> 4;                 // K-run half select
    const int woMask = Wo - 1;
    const int hwMask = (1 << hwShift) - 1;

    // decode this lane's output row (spatial position) with shifts only
    int m = mBase + col;
    bool mvalid = (m < Mtot);
    int mm = mvalid ? m : 0;
    int n  = mm >> hwShift;
    int rm = mm & hwMask;
    int oy = rm >> woShift;
    int ox = rm & woMask;
    int iy0 = oy * STRIDE - PAD;
    int ix0 = ox * STRIDE - PAD;
    const unsigned short* inN = in + (size_t)n * H * W * Cin;

    f32x8 acc[NB];
#pragma unroll
    for (int nb = 0; nb < NB; ++nb) {
        int c = (nTile0 + nb) * 16 + col;
        float bv = (c < Cout) ? bias[c] : 0.f;
#pragma unroll
        for (int r = 0; r < 8; ++r) acc[nb][r] = bv;
    }

    const unsigned short* wbase[NB];
#pragma unroll
    for (int nb = 0; nb < NB; ++nb)
        wbase[nb] = STAGE ? (sB  + ((size_t)nb * kChunks * 32 + lane) * 16)
                          : (wpk + ((size_t)(nTile0 + nb) * kChunks * 32 + lane) * 16);

    int kc = 0;
#pragma unroll
    for (int cell = 0; cell < KH * KW; ++cell) {
        const int kh = cell / KW;              // compile-time constants
        const int kw = cell - kh * KW;
        int iy = iy0 + kh;
        int ix = ix0 + kw;
        bool inb = mvalid && iy >= 0 && iy < H && ix >= 0 && ix < W;
        const unsigned short* src = inN + ((size_t)(iy * W + ix)) * Cin;
        for (int cc = 0; cc < cc32; ++cc, ++kc) {
            Frag af;
#pragma unroll
            for (int h2 = 0; h2 < 2; ++h2) {
                int ci0 = cc * 32 + hi * 8 + h2 * 16;
                if constexpr (VEC8) {
                    u32x4 v = {0u, 0u, 0u, 0u};
                    if (inb && ci0 < Cin) v = *(const u32x4*)(src + ci0);
                    af.d[h2] = v;
                } else {
#pragma unroll
                    for (int j = 0; j < 8; ++j) {
                        int ci = ci0 + j;
                        unsigned short v = 0;
                        if (inb && ci < Cin) v = src[ci];
                        af.s[h2 * 8 + j] = (short)v;
                    }
                }
            }
#pragma unroll
            for (int nb = 0; nb < NB; ++nb) {
                Frag bf;
                const u32x4* wq = (const u32x4*)(wbase[nb] + (size_t)kc * 512);
                bf.d[0] = wq[0];
                bf.d[1] = wq[1];
                acc[nb] = __builtin_amdgcn_wmma_f32_16x16x32_bf16(
                    false, af.b, false, bf.b, (short)0, acc[nb], false, false);
            }
        }
    }

#pragma unroll
    for (int nb = 0; nb < NB; ++nb) {
        int coutIdx = (nTile0 + nb) * 16 + col;
        if (coutIdx < Cout) {
#pragma unroll
            for (int r = 0; r < 8; ++r) {
                int mr = mBase + hi * 8 + r;   // D layout: VGPR r, lanes>=16 hold M=r+8
                if (mr < Mtot) {
                    int n2  = mr >> hwShift;
                    int rm2 = mr & hwMask;
                    int oy2 = rm2 >> woShift;
                    int ox2 = rm2 & woMask;
                    float v = acc[nb][r];
                    if (RELU) v = fmaxf(v, 0.f);
                    out[(((size_t)n2 * Ho + oy2) * Wo + ox2) * ldC + cBase + coutIdx] = f2bf(v);
                }
            }
        }
    }
}

// ---------------- 3x3/s2 maxpool, pad (0,1)(0,1) ceil, NHWC bf16 ----------------
__global__ void maxpool3x3(const unsigned short* __restrict__ in, unsigned short* __restrict__ out,
                           int N, int C, int H, int W, int Ho, int Wo,
                           int cShift, int woShift, int hoShift) {
    long idx = (long)blockIdx.x * blockDim.x + threadIdx.x;
    long total = (long)N * Ho * Wo * C;
    if (idx >= total) return;
    int c = (int)idx & (C - 1);
    long t = idx >> cShift;
    int ox = (int)t & (Wo - 1); t >>= woShift;
    int oy = (int)t & (Ho - 1);
    int n  = (int)(t >> hoShift);
    float m = -3.4e38f;
    for (int dy = 0; dy < 3; ++dy) {
        int iy = 2 * oy + dy;
        if (iy >= H) break;
        for (int dx = 0; dx < 3; ++dx) {
            int ix = 2 * ox + dx;
            if (ix >= W) break;
            float v = bf2f(in[(((size_t)n * H + iy) * W + ix) * C + c]);
            m = fmaxf(m, v);
        }
    }
    out[idx] = f2bf(m);
}

// ---------------- anchor decode (NC=1) ----------------
__global__ void resolve_kernel(const unsigned short* __restrict__ det, float* __restrict__ out,
                               int B, int A, int GW) {
    int idx = blockIdx.x * blockDim.x + threadIdx.x;
    if (idx >= B * A) return;
    int a = idx % A;
    int s = a % 6;
    int cell = a / 6;
    int gx = cell % GW;
    int gy = cell / GW;
    const float SW[6] = {6.f, 12.f, 18.f, 18.f, 20.f, 30.f};
    const float SH[6] = {5.f, 10.f, 10.f, 18.f, 24.f, 15.f};
    float ax = 8.f + 16.f * gx, ay = 8.f + 16.f * gy;
    float aw = SW[s], ah = SH[s];
    const unsigned short* y = det + (size_t)idx * 6;
    float score = 1.f / (1.f + __expf(-bf2f(y[1])));
    float cx = ax + aw * bf2f(y[2]);
    float cy = ay + ah * bf2f(y[3]);
    float w  = aw * __expf(bf2f(y[4]));
    float h  = ah * __expf(bf2f(y[5]));
    float* o = out + (size_t)idx * 7;
    o[0] = cx - 0.5f * w;
    o[1] = cy - 0.5f * h;
    o[2] = cx + 0.5f * w;
    o[3] = cy + 0.5f * h;
    o[4] = score;
    o[5] = 0.f;   // argmax over 1 class
    o[6] = 1.f;   // softmax over 1 class
}

static inline int ilog2(int v) { int s = 0; while ((1 << s) < v) ++s; return s; }

// ---------------- host orchestration ----------------
extern "C" void kernel_launch(void* const* d_in, const int* in_sizes, int n_in,
                              void* d_out, int out_size, void* d_ws, size_t ws_size,
                              hipStream_t stream) {
    (void)in_sizes; (void)n_in; (void)out_size; (void)ws_size;
    const float* x = (const float*)d_in[0];
    char* ws = (char*)d_ws;
    const size_t MiB = (size_t)1 << 20;
    unsigned short* A  = (unsigned short*)(ws);                 //  64 MiB activations
    unsigned short* Bb = (unsigned short*)(ws + 64 * MiB);      // 128 MiB activations
    unsigned short* S  = (unsigned short*)(ws + 192 * MiB);     //   8 MiB squeeze temp
    char*           Wb = ws + 200 * MiB;                        //   8 MiB packed weights
    unsigned short* D  = (unsigned short*)(ws + 208 * MiB);     //   2 MiB det head out

    { // input convert
        long total = 16L * 3 * 512 * 512;
        nchw_to_nhwc<<<(int)((total + 255) / 256), 256, 0, stream>>>(x, A, 16, 3, 512, 512);
    }

    size_t wcur = 0;
    int li = 0;
    // kind: 0=conv1(3x3 s2 p1, scalar gather), 1=1x1 relu, 2=3x3 p1 relu, 3=3x3 p1 no relu (direct B)
    auto conv = [&](int kind, const unsigned short* in, unsigned short* outp,
                    int H, int W, int Cin, int Ho, int Wo, int Cout, int ldC, int cBase) {
        const float* w = (const float*)d_in[1 + 2 * li];
        const float* b = (const float*)d_in[2 + 2 * li];
        ++li;
        int KH = (kind == 1) ? 1 : 3, KW = KH;
        int cc32 = (Cin + 31) / 32;
        int kChunks = KH * KW * cc32;
        int tiles = (Cout + 15) / 16;
        int NB = (tiles % 4 == 0) ? 4 : (tiles % 2 == 0) ? 2 : 1;
        int nTilesAlloc = ((tiles + NB - 1) / NB) * NB;
        unsigned short* wpk = (unsigned short*)(Wb + wcur);
        wcur += (size_t)nTilesAlloc * kChunks * 512 * sizeof(unsigned short);
        int ptotal = nTilesAlloc * kChunks * 512;
        pack_weights<<<(ptotal + 255) / 256, 256, 0, stream>>>(w, wpk, Cin, Cout, KH, KW, nTilesAlloc);
        int Mtot = 16 * Ho * Wo;
        dim3 grid((Mtot + 127) / 128, nTilesAlloc / NB);
        int woS = ilog2(Wo), hwS = ilog2(Ho * Wo);
        size_t smem = (size_t)kChunks * NB * 1024;   // bytes of staged packed-B (<=72KB)
        switch (kind * 8 + NB) {
        case 0 * 8 + 4: conv_wmma<3,3,2,1,true ,false,4,true ><<<grid,256,smem,stream>>>(in,wpk,b,outp,16,H,W,Cin,Ho,Wo,Cout,ldC,cBase,woS,hwS); break;
        case 1 * 8 + 1: conv_wmma<1,1,1,0,true ,true ,1,true ><<<grid,256,smem,stream>>>(in,wpk,b,outp,16,H,W,Cin,Ho,Wo,Cout,ldC,cBase,woS,hwS); break;
        case 1 * 8 + 2: conv_wmma<1,1,1,0,true ,true ,2,true ><<<grid,256,smem,stream>>>(in,wpk,b,outp,16,H,W,Cin,Ho,Wo,Cout,ldC,cBase,woS,hwS); break;
        case 1 * 8 + 4: conv_wmma<1,1,1,0,true ,true ,4,true ><<<grid,256,smem,stream>>>(in,wpk,b,outp,16,H,W,Cin,Ho,Wo,Cout,ldC,cBase,woS,hwS); break;
        case 2 * 8 + 4: conv_wmma<3,3,1,1,true ,true ,4,true ><<<grid,256,smem,stream>>>(in,wpk,b,outp,16,H,W,Cin,Ho,Wo,Cout,ldC,cBase,woS,hwS); break;
        case 3 * 8 + 1: conv_wmma<3,3,1,1,false,true ,1,false><<<grid,256,0,   stream>>>(in,wpk,b,outp,16,H,W,Cin,Ho,Wo,Cout,ldC,cBase,woS,hwS); break;
        default: break; // unused combination
        }
    };
    auto pool = [&](const unsigned short* in, unsigned short* outp, int H, int W, int C) {
        int Ho = (H - 2) / 2 + 1, Wo = (W - 2) / 2 + 1;
        long total = 16L * Ho * Wo * C;
        maxpool3x3<<<(int)((total + 255) / 256), 256, 0, stream>>>(
            in, outp, 16, C, H, W, Ho, Wo, ilog2(C), ilog2(Wo), ilog2(Ho));
    };
    auto fire = [&](const unsigned short* in, unsigned short* outp, int HW, int Cin, int sq, int e) {
        conv(1, in, S,    HW, HW, Cin, HW, HW, sq, sq,    0);   // squeeze 1x1
        conv(1, S,  outp, HW, HW, sq,  HW, HW, e,  2 * e, 0);   // expand 1x1 -> ch [0,e)
        conv(2, S,  outp, HW, HW, sq,  HW, HW, e,  2 * e, e);   // expand 3x3 -> ch [e,2e)
    };

    conv(0, A, Bb, 512, 512, 3, 256, 256, 64, 64, 0);   // conv1 + relu
    pool(Bb, A, 256, 256, 64);                          // -> 128x128x64   in A
    fire(A,  Bb, 128, 64,  16, 64);                     // f1 -> B (128ch)
    fire(Bb, A,  128, 128, 16, 64);                     // f2 -> A
    pool(A,  Bb, 128, 128, 128);                        // -> 64x64x128    in B
    fire(Bb, A,  64, 128, 32, 128);                     // f3 -> A (256ch)
    fire(A,  Bb, 64, 256, 32, 128);                     // f4 -> B
    pool(Bb, A,  64, 64, 256);                          // -> 32x32x256    in A
    fire(A,  Bb, 32, 256, 48, 192);                     // f5 -> B (384ch)
    fire(Bb, A,  32, 384, 48, 192);                     // f6 -> A
    fire(A,  Bb, 32, 384, 64, 256);                     // f7 -> B (512ch)
    fire(Bb, A,  32, 512, 64, 256);                     // f8 -> A
    conv(3, A, D, 32, 32, 512, 32, 32, 36, 36, 0);      // detection head (no relu)

    int rtotal = 16 * 6144;
    resolve_kernel<<<(rtotal + 255) / 256, 256, 0, stream>>>(D, (float*)d_out, 16, 6144, 32);
}
